// ReLuPCA_18665927868559
// MI455X (gfx1250) — matrix operand order repaired and checked by reading, not compiled, not run on hardware
//
#include <hip/hip_runtime.h>
#include <stdint.h>

// ---------------- problem constants ----------------
#define NB_  64
#define CC_  256
#define HH_  56
#define WW_  56
#define F_   128                         // mb*mb*(C/8) = 2*2*32
#define M_   401408                      // total / F_
#define TOT_ (NB_*CC_*HH_*WW_)           // 51,380,224
#define CHUNK_M    128                   // m-columns staged per block
#define NCHUNK     (M_/CHUNK_M)          // 3136 (exact)
#define LDS_STRIDE 136                   // u16 units per LDS row (272B = 128 + 8 pad)
#define COV_BLOCKS 256
#define MEAN_BLOCKS 784
#define QS_BLOCKS  98

// ---------------- workspace layout (bytes) ----------------
static constexpr size_t OFF_IMB   = 0;                              // bf16 im, [m][128]
static constexpr size_t SZ_IMB    = (size_t)M_ * F_ * 2;            // 102.7 MB
static constexpr size_t OFF_PROJ  = OFF_IMB + SZ_IMB;               // f32 imProj, [f][M]
static constexpr size_t SZ_PROJ   = (size_t)M_ * F_ * 4;            // 205.5 MB
static constexpr size_t OFF_COVP  = OFF_PROJ + SZ_PROJ;             // split-K cov partials
static constexpr size_t SZ_COVP   = (size_t)COV_BLOCKS * 16384 * 4; // 16 MB
static constexpr size_t OFF_MEANP = OFF_COVP + SZ_COVP;
static constexpr size_t SZ_MEANP  = (size_t)MEAN_BLOCKS * 128 * 4;
static constexpr size_t OFF_QSP   = OFF_MEANP + SZ_MEANP;
static constexpr size_t SZ_QSP    = (size_t)128 * QS_BLOCKS * 4;
static constexpr size_t OFF_COV   = OFF_QSP + SZ_QSP;               // 64 KB f32 cov (Jacobi in-place)
static constexpr size_t OFF_UF    = OFF_COV + 65536;                // 64 KB f32 eigenvectors V (=u)
static constexpr size_t OFF_UBF   = OFF_UF + 65536;                 // 32 KB bf16 u   [i][k]
static constexpr size_t OFF_UTBF  = OFF_UBF + 32768;                // 32 KB bf16 u^T [i][k]=u[k][i]
static constexpr size_t OFF_MN    = OFF_UTBF + 32768;               // 128 f32 per-feature mean
static constexpr size_t OFF_P     = OFF_MN + 512;                   // 128 f32 p = u^T mn
static constexpr size_t OFF_CF    = OFF_P + 512;                    // 128 f32 final per-row constant
static constexpr size_t OFF_MM    = OFF_CF + 512;                   // 2 u32 encoded min/max
static constexpr size_t OFF_QP    = OFF_MM + 64;                    // 3 f32: dmin, scale, 1/scale

// ---------------- types ----------------
typedef __attribute__((ext_vector_type(16))) __bf16 v16bf;
typedef __attribute__((ext_vector_type(8)))  float  v8f;

union BF16x16 { v16bf v; unsigned int u[8]; unsigned short h[16]; };
union F32x8   { v8f v; float f[8]; };

__device__ inline unsigned short f2bf(float f) {           // round-to-nearest-even
  unsigned int u = __float_as_uint(f);
  u += 0x7FFFu + ((u >> 16) & 1u);
  return (unsigned short)(u >> 16);
}
__device__ inline float bf2f(unsigned short h) { return __uint_as_float(((unsigned int)h) << 16); }

__device__ inline unsigned int encodeF(float f) {          // order-preserving f32 -> u32
  unsigned int u = __float_as_uint(f);
  return (u & 0x80000000u) ? ~u : (u | 0x80000000u);
}
__device__ inline float decodeF(unsigned int k) {
  unsigned int u = (k & 0x80000000u) ? (k & 0x7FFFFFFFu) : ~k;
  return __uint_as_float(u);
}

__device__ inline v8f wmma_bf16(const BF16x16& A, const BF16x16& B, v8f C) {
  return __builtin_amdgcn_wmma_f32_16x16x32_bf16(false, A.v, false, B.v, (short)0, C, false, false);
}

// ---------------- Tensor Data Mover staging (guarded, with manual fallback) ----
#if defined(__gfx1250__) && __has_builtin(__builtin_amdgcn_tensor_load_to_lds) && \
    __has_builtin(__builtin_amdgcn_s_wait_tensorcnt)
#define USE_TDM 1
#else
#define USE_TDM 0
#endif

#if USE_TDM
typedef unsigned int u32x4 __attribute__((ext_vector_type(4)));
typedef int          i32x4 __attribute__((ext_vector_type(4)));
typedef int          i32x8 __attribute__((ext_vector_type(8)));

// DMA a 128-row x 256B tile (64 DWORDs/row, dim0_stride=64) into LDS, with the
// TDM pad feature inserting 4 DWORDs of pad every 64 DWORDs -> 272B LDS pitch,
// matching LDS_STRIDE. Descriptor per CDNA5 ISA ch.8 (D# groups 0/1).
__device__ inline void tdm_load_tile(const void* gsrc, void* ldsDst) {
  unsigned lo = (unsigned)(size_t)ldsDst;               // LDS aperture addr[31:0]
  unsigned long long ga = (unsigned long long)(size_t)gsrc;
  u32x4 g0 = { 1u,                                      // count=1 (valid descriptor)
               lo,                                      // lds_addr
               (unsigned)ga,                            // global_addr[31:0]
               (unsigned)(ga >> 32) | 0x80000000u };    // global_addr[56:32] | type=2
  i32x8 g1 = { (int)((2u << 16) | (1u << 20) | (5u << 22) | (3u << 25)),
               //  data_size=4B   pad_en      pad_int=64dw  pad_amt=4dw
               (int)(64u << 16),                        // tensor_dim0 = 64 (lo16)
               (int)(128u << 16),                       // dim0 hi=0 | tensor_dim1=128 (lo16)
               (int)(64u << 16),                        // dim1 hi=0 | tile_dim0 = 64
               128,                                     // tile_dim1 = 128 | tile_dim2 = 0
               64,                                      // tensor_dim0_stride = 64 (lo32)
               0, 0 };
  i32x4 z4 = { 0, 0, 0, 0 };
#if __clang_major__ >= 23
  i32x8 z8 = { 0, 0, 0, 0, 0, 0, 0, 0 };
  __builtin_amdgcn_tensor_load_to_lds(g0, g1, z4, z4, z8, 0);
#else
  __builtin_amdgcn_tensor_load_to_lds(g0, g1, z4, z4, 0);
#endif
}
#endif

// ---------------- kernels ----------------
__global__ void k_init(unsigned int* mm) { mm[0] = 0xFFFFFFFFu; mm[1] = 0u; }

// x (NCHW) -> im bf16 in [m][128] layout (coalesced writes)
__global__ void k_gather(const float* __restrict__ x, unsigned short* __restrict__ imB) {
  long long t = (long long)blockIdx.x * blockDim.x + threadIdx.x;
  int f = (int)(t & 127);
  int m = (int)(t >> 7);
  int ct = f >> 2, hm = (f >> 1) & 1, wm = f & 1;
  int w2 = m % 28, h2 = (m / 28) % 28, g = (m / 784) % 8, n = m / 6272;
  int c = g * 32 + ct, h = h2 * 2 + hm, w = w2 * 2 + wm;
  size_t idx = ((size_t)(n * CC_ + c) * HH_ + h) * WW_ + w;
  imB[(size_t)m * F_ + f] = f2bf(x[idx]);
}

__global__ void k_mean_partial(const unsigned short* __restrict__ imB, float* __restrict__ partial) {
  __shared__ float red[512];
  int t = threadIdx.x, f = t & 127, mi = t >> 7;
  float s = 0.f;
  for (int it = 0; it < 128; ++it) {
    int m = blockIdx.x * 4 + mi + it * (MEAN_BLOCKS * 4);
    s += bf2f(imB[(size_t)m * F_ + f]);
  }
  red[t] = s; __syncthreads();
  if (t < 128) partial[blockIdx.x * 128 + t] = red[t] + red[t + 128] + red[t + 256] + red[t + 384];
}

__global__ void k_mean_final(const float* __restrict__ partial, float* __restrict__ mn) {
  int f = threadIdx.x; float s = 0.f;
  for (int b = 0; b < MEAN_BLOCKS; ++b) s += partial[b * 128 + f];
  mn[f] = s / (float)M_;
}

// Gram matrix (split-K, WMMA bf16, f32 accum).  A[i][k]=im[i][m], B[k][j]=im[j][m].
__global__ void __launch_bounds__(512) k_cov_partial(const unsigned short* __restrict__ imB,
                                                     float* __restrict__ covPart) {
  __shared__ __align__(16) unsigned short lds[CHUNK_M * LDS_STRIDE];
  int t = threadIdx.x, wave = t >> 5, lane = t & 31, col = lane & 15, hh = lane >> 4;
  int i0 = (wave & 7) * 16, jg = (wave >> 3) * 64;
  F32x8 acc[4];
  for (int a = 0; a < 4; ++a) for (int e = 0; e < 8; ++e) acc[a].f[e] = 0.f;

  for (int cc = blockIdx.x; cc < NCHUNK; cc += COV_BLOCKS) {
    __syncthreads();
#if USE_TDM
    if (wave == 0) {
      tdm_load_tile(imB + (size_t)cc * CHUNK_M * F_, lds);
      __builtin_amdgcn_s_wait_tensorcnt(0);
    }
#else
    {
      const uint4* src = (const uint4*)imB + (size_t)cc * CHUNK_M * 16;
      for (int e = t; e < CHUNK_M * 16; e += 512) {
        int row = e >> 4, c16 = e & 15;
        ((uint4*)lds)[row * 17 + c16] = src[row * 16 + c16];
      }
    }
#endif
    if (cc + COV_BLOCKS < NCHUNK) {  // prefetch next K-chunk -> global_prefetch_b8
      const char* nxt = (const char*)((const uint4*)imB + (size_t)(cc + COV_BLOCKS) * CHUNK_M * 16);
      __builtin_prefetch(nxt + t * 64, 0, 0);
    }
    __syncthreads();
    for (int kk = 0; kk < 4; ++kk) {
      int kb = kk * 32;
      BF16x16 A;
      #pragma unroll
      for (int e = 0; e < 16; ++e) {
        int v = e >> 1, s = e & 1;
        int K = 16 * hh + 2 * v + s;
        A.h[e] = lds[(kb + K) * LDS_STRIDE + i0 + col];
      }
      #pragma unroll
      for (int tt = 0; tt < 4; ++tt) {
        int j0 = jg + tt * 16;
        BF16x16 B;
        #pragma unroll
        for (int e = 0; e < 16; ++e)
          B.h[e] = lds[(kb + 16 * hh + e) * LDS_STRIDE + j0 + col];
        acc[tt].v = wmma_bf16(A, B, acc[tt].v);
      }
    }
  }
  float* dst = covPart + (size_t)blockIdx.x * 16384;
  for (int tt = 0; tt < 4; ++tt) {
    int j = jg + tt * 16 + col;
    #pragma unroll
    for (int r = 0; r < 8; ++r) dst[(i0 + 8 * hh + r) * 128 + j] = acc[tt].f[r];
  }
}

__global__ void k_cov_reduce(const float* __restrict__ covPart, const float* __restrict__ mn,
                             float* __restrict__ cov) {
  int e = blockIdx.x * 256 + threadIdx.x;  // < 16384
  float s = 0.f;
  for (int b = 0; b < COV_BLOCKS; ++b) s += covPart[(size_t)b * 16384 + e];
  cov[e] = s / (float)M_ - mn[e >> 7] * mn[e & 127];  // Gram/M - mn mn^T
}

// Parallel cyclic Jacobi eigensolver (single block). cov symmetric PSD -> u = V.
__global__ void k_jacobi(float* __restrict__ A, float* __restrict__ V,
                         const float* __restrict__ mn, float* __restrict__ p,
                         unsigned short* __restrict__ ubf, unsigned short* __restrict__ utbf) {
  __shared__ int idx[128], idx2[128], pA[64], qA[64];
  __shared__ float cA[64], sA[64];
  int t = threadIdx.x;
  for (int e = t; e < 16384; e += 128) V[e] = ((e >> 7) == (e & 127)) ? 1.f : 0.f;
  idx[t] = t;
  __syncthreads();
  for (int sweep = 0; sweep < 8; ++sweep) {
    for (int round = 0; round < 127; ++round) {
      if (t < 64) {
        int a = idx[t], b = idx[127 - t];
        int P = a < b ? a : b, Q = a < b ? b : a;
        pA[t] = P; qA[t] = Q;
        float app = A[P * 128 + P], aqq = A[Q * 128 + Q], apq = A[P * 128 + Q];
        float c = 1.f, s = 0.f;
        if (fabsf(apq) > 1e-20f) {
          float tau = (aqq - app) / (2.f * apq);
          float tn = (tau >= 0.f ? 1.f : -1.f) / (fabsf(tau) + sqrtf(1.f + tau * tau));
          c = rsqrtf(1.f + tn * tn);
          s = tn * c;
        }
        cA[t] = c; sA[t] = s;
      }
      __syncthreads();
      for (int wi = t; wi < 8192; wi += 128) {          // column rotations A, V
        int pr = wi >> 7, r = wi & 127;
        int P = pA[pr], Q = qA[pr]; float c = cA[pr], s = sA[pr];
        float ap = A[r * 128 + P], aq = A[r * 128 + Q];
        A[r * 128 + P] = c * ap - s * aq;
        A[r * 128 + Q] = s * ap + c * aq;
        float vp = V[r * 128 + P], vq = V[r * 128 + Q];
        V[r * 128 + P] = c * vp - s * vq;
        V[r * 128 + Q] = s * vp + c * vq;
      }
      __syncthreads();
      for (int wi = t; wi < 8192; wi += 128) {          // row rotations A
        int pr = wi >> 7, r = wi & 127;
        int P = pA[pr], Q = qA[pr]; float c = cA[pr], s = sA[pr];
        float ap = A[P * 128 + r], aq = A[Q * 128 + r];
        A[P * 128 + r] = c * ap - s * aq;
        A[Q * 128 + r] = s * ap + c * aq;
      }
      __syncthreads();
      idx2[t] = idx[t];
      __syncthreads();
      if (t >= 1) idx[t] = (t == 1) ? idx2[127] : idx2[t - 1];   // round-robin
      __syncthreads();
    }
  }
  for (int e = t; e < 16384; e += 128) {
    unsigned short b = f2bf(V[e]);
    ubf[e] = b;                                   // u[i][k]
    utbf[(e & 127) * 128 + (e >> 7)] = b;         // uT[k][i]
  }
  __syncthreads();
  float acc = 0.f;
  for (int k = 0; k < 128; ++k) acc += V[k * 128 + t] * mn[k];
  p[t] = acc;                                     // p = u^T mn
}

// imProj = u^T im - p ; track global min/max (order-independent atomics)
__global__ void __launch_bounds__(512) k_proj(const unsigned short* __restrict__ imB,
                                              const unsigned short* __restrict__ utbf,
                                              const float* __restrict__ p,
                                              float* __restrict__ imProj,
                                              unsigned int* __restrict__ mm) {
  __shared__ __align__(16) unsigned short lds[CHUNK_M * LDS_STRIDE];
  __shared__ float pL[128];
  int t = threadIdx.x, wave = t >> 5, lane = t & 31, col = lane & 15, hh = lane >> 4;
  int i0 = (wave & 7) * 16, jg = (wave >> 3) * 64;
  int m0 = blockIdx.x * CHUNK_M;
  if (t < 128) pL[t] = p[t];
#if USE_TDM
  if (wave == 0) {
    tdm_load_tile(imB + (size_t)blockIdx.x * CHUNK_M * F_, lds);
    __builtin_amdgcn_s_wait_tensorcnt(0);
  }
#else
  {
    const uint4* src = (const uint4*)imB + (size_t)blockIdx.x * CHUNK_M * 16;
    for (int e = t; e < CHUNK_M * 16; e += 512) {
      int row = e >> 4, c16 = e & 15;
      ((uint4*)lds)[row * 17 + c16] = src[row * 16 + c16];
    }
  }
#endif
  __syncthreads();
  F32x8 acc[4];
  for (int a = 0; a < 4; ++a) for (int e = 0; e < 8; ++e) acc[a].f[e] = 0.f;
  for (int kk = 0; kk < 4; ++kk) {
    int kb = kk * 32;
    BF16x16 A;
    #pragma unroll
    for (int v = 0; v < 8; ++v) {
      int K0 = 16 * hh + 2 * v;
      A.u[v] = *(const unsigned int*)(utbf + (i0 + col) * 128 + kb + K0);
    }
    #pragma unroll
    for (int tt = 0; tt < 4; ++tt) {
      int j0 = jg + tt * 16;
      BF16x16 B;
      #pragma unroll
      for (int v = 0; v < 8; ++v)
        B.u[v] = *(const unsigned int*)(lds + (j0 + col) * LDS_STRIDE + kb + 16 * hh + 2 * v);
      acc[tt].v = wmma_bf16(A, B, acc[tt].v);
    }
  }
  float lmin = 3.4e38f, lmax = -3.4e38f;
  for (int tt = 0; tt < 4; ++tt) {
    int j = jg + tt * 16 + col;
    #pragma unroll
    for (int r = 0; r < 8; ++r) {
      int row = i0 + 8 * hh + r;
      float v = acc[tt].f[r] - pL[row];
      lmin = fminf(lmin, v); lmax = fmaxf(lmax, v);
      imProj[(size_t)row * M_ + m0 + j] = v;
    }
  }
  atomicMin(&mm[0], encodeF(lmin));
  atomicMax(&mm[1], encodeF(lmax));
}

__global__ void k_qparams(const unsigned int* __restrict__ mm, float* __restrict__ qp) {
  float dmin = decodeF(mm[0]), dmax = decodeF(mm[1]);
  float scale = 254.f / (dmax - dmin);        // 2^8 - 2
  qp[0] = dmin; qp[1] = scale; qp[2] = 1.f / scale;
}

__global__ void k_qsum_partial(const float* __restrict__ imProj, const float* __restrict__ qp,
                               float* __restrict__ partial) {
  __shared__ float red[512];
  int f = blockIdx.y, t = threadIdx.x;
  float dmin = qp[0], scl = qp[1], inv = qp[2], s = 0.f;
  for (int it = 0; it < 8; ++it) {
    int m = blockIdx.x * 512 + t + it * (QS_BLOCKS * 512);
    float v = imProj[(size_t)f * M_ + m];
    s += rintf((v - dmin) * scl) * inv + dmin;   // quant-dequant
  }
  red[t] = s; __syncthreads();
  for (int off = 256; off > 0; off >>= 1) { if (t < off) red[t] += red[t + off]; __syncthreads(); }
  if (t == 0) partial[f * QS_BLOCKS + blockIdx.x] = red[0];
}

__global__ void k_cfinal(const float* __restrict__ partial, const float* __restrict__ mn,
                         const float* __restrict__ V, float* __restrict__ cf) {
  __shared__ float qr[128];
  int t = threadIdx.x; float s = 0.f;
  for (int b = 0; b < QS_BLOCKS; ++b) s += partial[t * QS_BLOCKS + b];
  qr[t] = s / (float)M_;
  __syncthreads();
  float a = 0.f;
  for (int k = 0; k < 128; ++k) a += V[t * 128 + k] * qr[k];
  cf[t] = mn[t] - a;       // out = u q + mn - u rowmean(q)
}

// out = relu( u * quant(imProj) + cf ), scattered back to NCHW.
// Staging fuses quantization, so it stays on the manual (compute) path.
__global__ void __launch_bounds__(512) k_recon(const float* __restrict__ imProj,
                                               const unsigned short* __restrict__ ubf,
                                               const float* __restrict__ qp,
                                               const float* __restrict__ cf,
                                               float* __restrict__ out) {
  __shared__ __align__(16) unsigned short lds[CHUNK_M * LDS_STRIDE];
  __shared__ float cL[128];
  int t = threadIdx.x, wave = t >> 5, lane = t & 31, col = lane & 15, hh = lane >> 4;
  int i0 = (wave & 7) * 16, jg = (wave >> 3) * 64;
  int m0 = blockIdx.x * CHUNK_M;
  if (t < 128) cL[t] = cf[t];
  float dmin = qp[0], scl = qp[1], inv = qp[2];
  for (int e = t; e < CHUNK_M * 128; e += 512) {     // stage + quantize + bf16
    int f = e >> 7, j = e & 127;
    float v = imProj[(size_t)f * M_ + m0 + j];
    float vq = rintf((v - dmin) * scl) * inv + dmin;
    lds[j * LDS_STRIDE + f] = f2bf(vq);
  }
  __syncthreads();
  F32x8 acc[4];
  for (int a = 0; a < 4; ++a) for (int e = 0; e < 8; ++e) acc[a].f[e] = 0.f;
  for (int kk = 0; kk < 4; ++kk) {
    int kb = kk * 32;
    BF16x16 A;
    #pragma unroll
    for (int v = 0; v < 8; ++v) {
      int K0 = 16 * hh + 2 * v;
      A.u[v] = *(const unsigned int*)(ubf + (i0 + col) * 128 + kb + K0);
    }
    #pragma unroll
    for (int tt = 0; tt < 4; ++tt) {
      int j0 = jg + tt * 16;
      BF16x16 B;
      #pragma unroll
      for (int v = 0; v < 8; ++v)
        B.u[v] = *(const unsigned int*)(lds + (j0 + col) * LDS_STRIDE + kb + 16 * hh + 2 * v);
      acc[tt].v = wmma_bf16(A, B, acc[tt].v);
    }
  }
  for (int tt = 0; tt < 4; ++tt) {
    int m = m0 + jg + tt * 16 + col;
    int w2 = m % 28, h2 = (m / 28) % 28, g = (m / 784) % 8, n = m / 6272;
    #pragma unroll
    for (int r = 0; r < 8; ++r) {
      int row = i0 + 8 * hh + r;
      float v = fmaxf(acc[tt].f[r] + cL[row], 0.f);
      int ct = row >> 2, hm = (row >> 1) & 1, wm = row & 1;
      int c = g * 32 + ct, h = h2 * 2 + hm, w = w2 * 2 + wm;
      out[((size_t)(n * CC_ + c) * HH_ + h) * WW_ + w] = v;
    }
  }
}

// ---------------- launch ----------------
extern "C" void kernel_launch(void* const* d_in, const int* in_sizes, int n_in,
                              void* d_out, int out_size, void* d_ws, size_t ws_size,
                              hipStream_t stream) {
  (void)in_sizes; (void)n_in; (void)out_size; (void)ws_size;
  const float* x = (const float*)d_in[0];
  float* out = (float*)d_out;
  char* ws = (char*)d_ws;

  unsigned short* imB  = (unsigned short*)(ws + OFF_IMB);
  float*          proj = (float*)(ws + OFF_PROJ);
  float*          covP = (float*)(ws + OFF_COVP);
  float*          mnP  = (float*)(ws + OFF_MEANP);
  float*          qsP  = (float*)(ws + OFF_QSP);
  float*          cov  = (float*)(ws + OFF_COV);
  float*          uF   = (float*)(ws + OFF_UF);
  unsigned short* ubf  = (unsigned short*)(ws + OFF_UBF);
  unsigned short* utbf = (unsigned short*)(ws + OFF_UTBF);
  float*          mn   = (float*)(ws + OFF_MN);
  float*          p    = (float*)(ws + OFF_P);
  float*          cf   = (float*)(ws + OFF_CF);
  unsigned int*   mm   = (unsigned int*)(ws + OFF_MM);
  float*          qp   = (float*)(ws + OFF_QP);

  k_init        <<<1, 1, 0, stream>>>(mm);
  k_gather      <<<TOT_ / 512, 512, 0, stream>>>(x, imB);
  k_mean_partial<<<MEAN_BLOCKS, 512, 0, stream>>>(imB, mnP);
  k_mean_final  <<<1, 128, 0, stream>>>(mnP, mn);
  k_cov_partial <<<COV_BLOCKS, 512, 0, stream>>>(imB, covP);
  k_cov_reduce  <<<64, 256, 0, stream>>>(covP, mn, cov);
  k_jacobi      <<<1, 128, 0, stream>>>(cov, uF, mn, p, ubf, utbf);
  k_proj        <<<NCHUNK, 512, 0, stream>>>(imB, utbf, p, proj, mm);
  k_qparams     <<<1, 1, 0, stream>>>(mm, qp);
  k_qsum_partial<<<dim3(QS_BLOCKS, 128), 512, 0, stream>>>(proj, qp, qsP);
  k_cfinal      <<<1, 128, 0, stream>>>(qsP, mn, uF, cf);
  k_recon       <<<NCHUNK, 512, 0, stream>>>(proj, ubf, qp, cf, out);
}
